// PatchReader2ConvLayerRetrieve_89653147336984
// MI455X (gfx1250) — compile-verified, matrix operand order checked by reading, not run on hardware
//
#include <hip/hip_runtime.h>
#include <hip/hip_bf16.h>
#include <stdint.h>

// ---------------- CDNA5 WMMA types ----------------
typedef __attribute__((ext_vector_type(16))) __bf16 v16bf;
typedef __attribute__((ext_vector_type(8)))  float  v8f;

#define NEG_SLOPE 0.01f
#define EPS_N 1e-5f

// Native hardware f32 -> bf16 conversion helpers
__device__ __forceinline__ unsigned short bf16u(float f) {
    return __builtin_bit_cast(unsigned short, (__bf16)f);
}
__device__ __forceinline__ unsigned int pack_bf16x2(float lo, float hi) {
    return (unsigned int)bf16u(lo) | ((unsigned int)bf16u(hi) << 16);
}
__device__ __forceinline__ float leaky(float v) {
    return v >= 0.f ? v : NEG_SLOPE * v;
}

struct U8 { unsigned int u[8]; };   // 32B == sizeof(v16bf)

// ---------------------------------------------------------------------------
// GEMM: C[M x N] = (A[M x K] @ Bt[N x K]^T) * row_scale[m]
//   A  : bf16, row-major [M][K]          (M%64==0, K%64==0)
//   Bt : bf16, row-major [N_pad][K]      (N_pad%128==0, zero-padded rows)
//   C  : f32,  row-major [M][N]
// Block: 256 threads = 8 waves.  Block tile 64(M) x 128(N), K stepped by 64.
// Tiles staged via CDNA5 async memory->LDS copies (ASYNCcnt), fragments via
// ds_load_b128, math via v_wmma_f32_16x16x32_bf16 (8 WMMAs / K-step).
// ---------------------------------------------------------------------------
__global__ __launch_bounds__(256) void gemm_bf16t_wmma(
    const unsigned short* __restrict__ A, const unsigned short* __restrict__ Bt,
    float* __restrict__ C, const float* __restrict__ row_scale,
    int M, int N, int K)
{
    // packed bf16 pairs along K (32 uints per 64-K row)
    __shared__ unsigned int Asu[64 * 32];    // 8 KB  [m][kpair]
    __shared__ unsigned int Bsu[128 * 32];   // 16 KB [n][kpair]

    const int t    = threadIdx.x;
    const int wid  = t >> 5;
    const int lane = t & 31;
    const int hi   = lane >> 4;
    const int lm   = lane & 15;
    const int m_base = blockIdx.y * 64;
    const int n_base = blockIdx.x * 128;
    const int n_loc  = wid * 16 + lm;

    v8f acc[4];
    #pragma unroll
    for (int r = 0; r < 4; ++r) acc[r] = (v8f){0.f,0.f,0.f,0.f,0.f,0.f,0.f,0.f};

    for (int k0 = 0; k0 < K; k0 += 64) {
        // ---- prefetch next K tile (global_prefetch_b8) ----
        if (k0 + 64 < K) {
            __builtin_prefetch(A  + (size_t)(m_base + (t >> 2)) * K + (k0 + 64) + ((t & 3) << 4), 0, 0);
            __builtin_prefetch(Bt + (size_t)(n_base + (t >> 1)) * K + (k0 + 64) + ((t & 1) << 5), 0, 0);
        }

        // ---- async copy A tile: 64 rows x 128B  (2 x 256 lanes x 16B) ----
        #pragma unroll
        for (int i = 0; i < 2; ++i) {
            int idx = i * 256 + t;
            int row = idx >> 3;            // 0..63
            int ch  = idx & 7;             // 16B chunk within 128B row slice
            const unsigned short* gp = A + (size_t)(m_base + row) * K + k0 + (ch << 3);
            unsigned int lo = (unsigned int)(size_t)&Asu[row * 32 + (ch << 2)];
            asm volatile("global_load_async_to_lds_b128 %0, %1, off"
                         :: "v"(lo), "v"(gp) : "memory");
        }
        // ---- async copy B tile: 128 rows x 128B  (4 x 256 lanes x 16B) ----
        #pragma unroll
        for (int i = 0; i < 4; ++i) {
            int idx = i * 256 + t;
            int row = idx >> 3;            // 0..127
            int ch  = idx & 7;
            const unsigned short* gp = Bt + (size_t)(n_base + row) * K + k0 + (ch << 3);
            unsigned int lo = (unsigned int)(size_t)&Bsu[row * 32 + (ch << 2)];
            asm volatile("global_load_async_to_lds_b128 %0, %1, off"
                         :: "v"(lo), "v"(gp) : "memory");
        }
        asm volatile("s_wait_asynccnt 0x0" ::: "memory");
        __syncthreads();

        // ---- 8 WMMAs: 2 K-halves x 4 M sub-tiles (B frag reused) ----
        #pragma unroll
        for (int kh = 0; kh < 2; ++kh) {
            U8 bu;
            #pragma unroll
            for (int v = 0; v < 8; ++v) {
                // pair index: k = 32*kh + 16*(v/4) + 8*hi + 2*(v&3)
                int kp = kh * 16 + ((v >> 2) << 3) + (hi << 2) + (v & 3);
                bu.u[v] = Bsu[n_loc * 32 + kp];
            }
            v16bf b = __builtin_bit_cast(v16bf, bu);
            #pragma unroll
            for (int r = 0; r < 4; ++r) {
                U8 au;
                #pragma unroll
                for (int v = 0; v < 8; ++v) {
                    int kp = kh * 16 + ((v >> 2) << 3) + (hi << 2) + (v & 3);
                    au.u[v] = Asu[(r * 16 + lm) * 32 + kp];
                }
                v16bf a = __builtin_bit_cast(v16bf, au);
                acc[r] = __builtin_amdgcn_wmma_f32_16x16x32_bf16(
                    false, a, false, b, (short)0, acc[r], false, false);
            }
        }
        __syncthreads();
    }

    // ---- epilogue: D element (M = m_base + 16r + 8*hi + q, N = n_loc) ----
    int n_g = n_base + n_loc;           // wave-uniform in/out of range
    if (n_g < N) {
        #pragma unroll
        for (int r = 0; r < 4; ++r) {
            #pragma unroll
            for (int q = 0; q < 8; ++q) {
                int m_g = m_base + r * 16 + hi * 8 + q;
                float s = row_scale ? row_scale[m_g] : 1.f;
                C[(size_t)m_g * N + n_g] = acc[r][q] * s;
            }
        }
    }
}

// ---------------------------------------------------------------------------
// f32 -> bf16 flat convert (2 floats -> 1 packed uint per thread)
// ---------------------------------------------------------------------------
__global__ void convert_bf16_kernel(const float* __restrict__ in,
                                    unsigned short* __restrict__ out, long long n2)
{
    long long i = (long long)blockIdx.x * blockDim.x + threadIdx.x;
    if (i < n2) {
        float2 v = ((const float2*)in)[i];
        ((unsigned int*)out)[i] = pack_bf16x2(v.x, v.y);
    }
}

// ---------------------------------------------------------------------------
// W[K x N] f32 -> Wt[N_pad x K] bf16 (tiled transpose, zero-padded rows)
// grid: (K/32, Npad/32), block 256
// ---------------------------------------------------------------------------
__global__ __launch_bounds__(256) void transpose_bf16_kernel(
    const float* __restrict__ W, unsigned short* __restrict__ Wt,
    int K, int N, int Npad)
{
    __shared__ float tile[32][33];
    int kb = blockIdx.x * 32, nb = blockIdx.y * 32;
    int tx = threadIdx.x & 31, ty = threadIdx.x >> 5;   // ty: 0..7
    #pragma unroll
    for (int j = 0; j < 4; ++j) {
        int k = kb + ty + 8 * j, n = nb + tx;
        tile[ty + 8 * j][tx] = (n < N) ? W[(size_t)k * N + n] : 0.f;
    }
    __syncthreads();
    #pragma unroll
    for (int j = 0; j < 4; ++j) {
        int n = nb + ty + 8 * j, k = kb + tx;
        if (n < Npad) Wt[(size_t)n * K + k] = bf16u(tile[tx][ty + 8 * j]);
    }
}

// ---------------------------------------------------------------------------
// Degree accumulation + rsqrt(max(deg,1))
// ---------------------------------------------------------------------------
__global__ void deg_kernel(const int* __restrict__ src, const int* __restrict__ dst,
                           float* degO, float* degI, int E)
{
    int i = blockIdx.x * blockDim.x + threadIdx.x;
    if (i < E) {
        atomicAdd(&degO[src[i]], 1.f);
        atomicAdd(&degI[dst[i]], 1.f);
    }
}

__global__ void rsqrt_kernel(float* p, int n)
{
    int i = blockIdx.x * blockDim.x + threadIdx.x;
    if (i < n) p[i] = rsqrtf(fmaxf(p[i], 1.f));
}

// ---------------------------------------------------------------------------
// Edge scatter: agg[dst] += h[src] * ew   (float4 gather, 4x f32 atomic add)
// ---------------------------------------------------------------------------
__global__ void scatter_kernel(const float* __restrict__ h, const int* __restrict__ src,
                               const int* __restrict__ dst, const float* __restrict__ ew,
                               float* __restrict__ agg, int E, int C)
{
    int C4 = C >> 2;
    long long gid = (long long)blockIdx.x * blockDim.x + threadIdx.x;
    if (gid >= (long long)E * C4) return;
    int e  = (int)(gid / C4);
    int c4 = (int)(gid % C4);
    int s = src[e], d = dst[e];
    float w = ew[e];
    const float4 v = *(const float4*)(h + (size_t)s * C + c4 * 4);
    float* ap = agg + (size_t)d * C + c4 * 4;
    atomicAdd(ap + 0, v.x * w);
    atomicAdd(ap + 1, v.y * w);
    atomicAdd(ap + 2, v.z * w);
    atomicAdd(ap + 3, v.w * w);
}

// ---------------------------------------------------------------------------
// Fused: conv-out = agg * rsqrt(deg_in); leaky; GraphNorm.  Emits bf16 (next
// GEMM's A operand).  One block per graph, one thread per channel.
// ---------------------------------------------------------------------------
__global__ void graphnorm_kernel(const float* __restrict__ in, const float* __restrict__ rs_in,
                                 const float* __restrict__ alpha, const float* __restrict__ gamma,
                                 const float* __restrict__ beta, unsigned short* __restrict__ out,
                                 int C, int nper)
{
    __shared__ float sv[64 * 128];       // nper<=64, C<=128
    int g = blockIdx.x, c = threadIdx.x;
    float sum = 0.f;
    for (int i = 0; i < nper; ++i) {
        int row = g * nper + i;
        float v = in[(size_t)row * C + c] * rs_in[row];
        v = leaky(v);
        sv[i * C + c] = v;
        sum += v;
    }
    float mean = sum / (float)nper;
    float a = alpha[c], gm = gamma[c], bt = beta[c];
    float var = 0.f;
    for (int i = 0; i < nper; ++i) {
        float sub = sv[i * C + c] - a * mean;
        sv[i * C + c] = sub;
        var += sub * sub;
    }
    float inv = rsqrtf(var / (float)nper + EPS_N);
    for (int i = 0; i < nper; ++i)
        out[(size_t)(g * nper + i) * C + c] = bf16u(gm * sv[i * C + c] * inv + bt);
}

// ---------------------------------------------------------------------------
// Fused: leaky + InstanceNorm1d(affine=False) per row; f32 in, bf16 out.
// ---------------------------------------------------------------------------
__global__ __launch_bounds__(256) void instnorm_kernel(
    const float* __restrict__ x, unsigned short* __restrict__ out, int F)
{
    __shared__ float buf[2048];
    __shared__ float red1[256], red2[256];
    int row = blockIdx.x, t = threadIdx.x;
    float s = 0.f, ss = 0.f;
    for (int i = t; i < F; i += 256) {
        float v = leaky(x[(size_t)row * F + i]);
        buf[i] = v; s += v; ss += v * v;
    }
    red1[t] = s; red2[t] = ss;
    __syncthreads();
    for (int off = 128; off > 0; off >>= 1) {
        if (t < off) { red1[t] += red1[t + off]; red2[t] += red2[t + off]; }
        __syncthreads();
    }
    float mean = red1[0] / (float)F;
    float var  = red2[0] / (float)F - mean * mean;
    float inv  = rsqrtf(var + EPS_N);
    for (int i = t; i < F; i += 256)
        out[(size_t)row * F + i] = bf16u((buf[i] - mean) * inv);
}

// ---------------------------------------------------------------------------
extern "C" void kernel_launch(void* const* d_in, const int* in_sizes, int n_in,
                              void* d_out, int out_size, void* d_ws, size_t ws_size,
                              hipStream_t stream)
{
    const float* x    = (const float*)d_in[0];
    const float* ew   = (const float*)d_in[1];
    const int*   src  = (const int*)d_in[2];
    const int*   dst  = (const int*)d_in[3];
    const float* W1   = (const float*)d_in[4];
    const float* W2   = (const float*)d_in[5];
    const float* a1   = (const float*)d_in[6];
    const float* g1   = (const float*)d_in[7];
    const float* b1   = (const float*)d_in[8];
    const float* a2   = (const float*)d_in[9];
    const float* g2   = (const float*)d_in[10];
    const float* b2   = (const float*)d_in[11];
    const float* Win  = (const float*)d_in[12];
    const float* Whid = (const float*)d_in[13];
    const float* Wcls = (const float*)d_in[14];

    const int in_feats = 64;
    const int Nn   = in_sizes[0] / in_feats;     // 262144 nodes
    const int E    = in_sizes[1];                // 2M edges
    const int Cd   = in_sizes[4] / in_feats;     // conv_dim = 128
    const int Co   = in_sizes[5] / Cd;           // out_conv = 96
    const int Bg   = 4096;                       // batch_size (reference)
    const int nper = Nn / Bg;                    // 64 nodes per graph
    const int in_mlp = Co * nper;                // 6144
    const int hid  = in_sizes[12] / in_mlp;      // 2048
    const int hid4 = in_sizes[13] / hid;         // 512
    const int ofea = in_sizes[14] / hid4;        // 256
    const int CdP  = (Cd  + 127) & ~127;         // 128
    const int CoP  = (Co  + 127) & ~127;         // 128
    const int ofeaP= (ofea+ 127) & ~127;         // 256

    // workspace: [degO Nn][degI Nn][R0 Nn*Cd f32][R1 Nn*Cd f32][transposed weights]
    float* degO = (float*)d_ws;
    float* degI = degO + Nn;
    float* R0   = degI + Nn;
    float* R1   = R0 + (size_t)Nn * Cd;
    unsigned short* W1t   = (unsigned short*)(R1 + (size_t)Nn * Cd);
    unsigned short* W2t   = W1t  + (size_t)CdP * in_feats;
    unsigned short* Wint  = W2t  + (size_t)CoP * Cd;
    unsigned short* Whidt = Wint + (size_t)hid * in_mlp;
    unsigned short* Wclst = Whidt + (size_t)hid4 * hid;

    // bf16 activation buffers overlaid on dead f32 regions
    unsigned short* xb    = (unsigned short*)R1;   // Nn*64   (dead after gemm1)
    unsigned short* h1b   = (unsigned short*)R0;   // Nn*Cd   (dead after gemm2)
    unsigned short* xmlpb = (unsigned short*)R1;   // Nn*Co   (dead after MLP1)
    unsigned short* z1b   = (unsigned short*)R1;   // Bg*hid  (dead after MLP2)
    unsigned short* z2b   = (unsigned short*)R1;   // Bg*hid4 (dead after MLP3)

    // --- weight transposes (bf16, [N_pad][K]) ---
    transpose_bf16_kernel<<<dim3(in_feats/32, CdP/32), 256, 0, stream>>>(W1,  W1t,  in_feats, Cd,  CdP);
    transpose_bf16_kernel<<<dim3(Cd/32,      CoP/32), 256, 0, stream>>>(W2,  W2t,  Cd,  Co,  CoP);
    transpose_bf16_kernel<<<dim3(in_mlp/32,  hid/32), 256, 0, stream>>>(Win, Wint, in_mlp, hid, hid);
    transpose_bf16_kernel<<<dim3(hid/32,    hid4/32), 256, 0, stream>>>(Whid, Whidt, hid, hid4, hid4);
    transpose_bf16_kernel<<<dim3(hid4/32,  ofeaP/32), 256, 0, stream>>>(Wcls, Wclst, hid4, ofea, ofeaP);

    // --- degrees -> rsqrt(max(deg,1)) ---
    hipMemsetAsync(degO, 0, sizeof(float) * 2 * (size_t)Nn, stream);
    deg_kernel<<<(E + 255) / 256, 256, 0, stream>>>(src, dst, degO, degI, E);
    rsqrt_kernel<<<(2 * Nn + 255) / 256, 256, 0, stream>>>(degO, 2 * Nn);

    // --- x -> bf16 ---
    {
        long long n2 = (long long)Nn * in_feats / 2;
        convert_bf16_kernel<<<(unsigned)((n2 + 255) / 256), 256, 0, stream>>>(x, xb, n2);
    }

    // --- conv1: h0 = (x @ W1) * rs_out ---
    gemm_bf16t_wmma<<<dim3(CdP / 128, Nn / 64), 256, 0, stream>>>(
        xb, W1t, R0, degO, Nn, Cd, in_feats);
    hipMemsetAsync(R1, 0, sizeof(float) * (size_t)Nn * Cd, stream);
    {
        long long tot = (long long)E * (Cd / 4);
        scatter_kernel<<<(unsigned)((tot + 255) / 256), 256, 0, stream>>>(
            R0, src, dst, ew, R1, E, Cd);
    }
    graphnorm_kernel<<<Bg, Cd, 0, stream>>>(R1, degI, a1, g1, b1, h1b, Cd, nper);

    // --- conv2: h2raw = (h1 @ W2) * rs_out ---
    gemm_bf16t_wmma<<<dim3(CoP / 128, Nn / 64), 256, 0, stream>>>(
        h1b, W2t, R1, degO, Nn, Co, Cd);
    hipMemsetAsync(R0, 0, sizeof(float) * (size_t)Nn * Co, stream);
    {
        long long tot = (long long)E * (Co / 4);
        scatter_kernel<<<(unsigned)((tot + 255) / 256), 256, 0, stream>>>(
            R1, src, dst, ew, R0, E, Co);
    }
    graphnorm_kernel<<<Bg, Co, 0, stream>>>(R0, degI, a2, g2, b2, xmlpb, Co, nper);
    // xmlpb == Xmlp viewed as bf16 [Bg x in_mlp] (row-major reshape is free)

    // --- MLP ---
    gemm_bf16t_wmma<<<dim3(hid / 128, Bg / 64), 256, 0, stream>>>(
        xmlpb, Wint, R0, nullptr, Bg, hid, in_mlp);
    instnorm_kernel<<<Bg, 256, 0, stream>>>(R0, z1b, hid);

    gemm_bf16t_wmma<<<dim3(hid4 / 128, Bg / 64), 256, 0, stream>>>(
        z1b, Whidt, R0, nullptr, Bg, hid4, hid);
    instnorm_kernel<<<Bg, 256, 0, stream>>>(R0, z2b, hid4);

    gemm_bf16t_wmma<<<dim3(ofeaP / 128, Bg / 64), 256, 0, stream>>>(
        z2b, Wclst, (float*)d_out, nullptr, Bg, ofea, hid4);
}
